// LGAGenerator_79577154060655
// MI455X (gfx1250) — matrix operand order repaired and checked by reading, not compiled, not run on hardware
//
#include <hip/hip_runtime.h>
#include <math.h>

// ---------------- CDNA5 WMMA types ----------------
typedef __attribute__((ext_vector_type(16))) __bf16 v16bf;
typedef __attribute__((ext_vector_type(8)))  float  v8f;

union FragBF16 { uint4 u[2]; v16bf v; };  // 32 bytes = 16 bf16 per lane

#define D_IN  512
#define D_HID 256

// ---------------- degree / norm ----------------
__global__ void k_deg_init(float* __restrict__ deg, int n) {
    int i = blockIdx.x * blockDim.x + threadIdx.x;
    if (i < n) deg[i] = 1.0f;                     // self-loop
}

__global__ void k_deg_scatter(const int* __restrict__ dst, float* __restrict__ deg, int e) {
    int i = blockIdx.x * blockDim.x + threadIdx.x;
    if (i < e) atomicAdd(&deg[dst[i]], 1.0f);
}

__global__ void k_rsqrt_inplace(float* __restrict__ d, int n) {
    int i = blockIdx.x * blockDim.x + threadIdx.x;
    if (i < n) d[i] = rsqrtf(d[i]);               // deg >= 1 always
}

// ---------------- fp32 -> bf16 convert ----------------
__global__ void k_cvt_bf16(const float* __restrict__ in, __bf16* __restrict__ out, long n) {
    long i = (long)blockIdx.x * blockDim.x + threadIdx.x;
    if (i < n) out[i] = (__bf16)in[i];
}

// Pack W [K x N, row-major f32] into WMMA B-fragment order (bf16):
//   out[((kt*NT + nt)*32 + lane)*16 + j]
//   lane = (n%16) | (k%32 >= 16 ? 16 : 0),  j = k%16   (contiguous k order, same as the A load)
__global__ void k_pack_w(const float* __restrict__ W, __bf16* __restrict__ Wp, int K, int N) {
    int i = blockIdx.x * blockDim.x + threadIdx.x;
    if (i >= K * N) return;
    int k = i / N, n = i % N;
    int kt = k >> 5, kk = k & 31, nt = n >> 4, nn = n & 15;
    int lane = nn | (kk & 16);
    int j = kk & 15;
    int NT = N >> 4;
    Wp[(((long)(kt * NT + nt)) * 32 + lane) * 16 + j] = (__bf16)W[i];
}

// ---------------- bf16 WMMA GEMM: C[M x 256] = A[M x K] * Bpacked ----------------
// One block = one 16-row M tile. 8 waves; wave w computes N-tiles w and w+8.
__global__ void __launch_bounds__(256)
k_gemm_bf16(const __bf16* __restrict__ A, const __bf16* __restrict__ Bp,
            float* __restrict__ C, int K) {
    const int lane = threadIdx.x & 31;
    const int wave = threadIdx.x >> 5;
    const int hi   = lane >> 4;              // 0 | 1 (lane group)
    const int m0   = blockIdx.x << 4;
    const int KT   = K >> 5;
    const __bf16* arow = A + (long)(m0 + (lane & 15)) * K;

    const int nt0 = wave;
    const int nt1 = wave + 8;
    v8f acc0 = {};
    v8f acc1 = {};

    for (int kt = 0; kt < KT; ++kt) {
        // A fragment (contiguous k order, same as packed B): lane<16 -> K {0..15}; lane>=16 -> K {16..31}
        FragBF16 a;
        const int kb = (kt << 5) + (hi << 4);
        a.u[0] = *(const uint4*)(arow + kb);
        a.u[1] = *(const uint4*)(arow + kb + 8);

        const __bf16* bbase = Bp + (((long)kt << 4) << 9) + ((long)lane << 4);
        FragBF16 b0, b1;
        const __bf16* p0 = bbase + ((long)nt0 << 9);
        const __bf16* p1 = bbase + ((long)nt1 << 9);
        b0.u[0] = *(const uint4*)(p0);
        b0.u[1] = *(const uint4*)(p0 + 8);
        b1.u[0] = *(const uint4*)(p1);
        b1.u[1] = *(const uint4*)(p1 + 8);

        acc0 = __builtin_amdgcn_wmma_f32_16x16x32_bf16(false, a.v, false, b0.v,
                                                       (short)0, acc0, false, false);
        acc1 = __builtin_amdgcn_wmma_f32_16x16x32_bf16(false, a.v, false, b1.v,
                                                       (short)0, acc1, false, false);
    }

    // D layout: vgpr r -> row r + 8*hi, col = lane&15
    float* c0 = C + ((long)m0 + (hi << 3)) * D_HID + (nt0 << 4) + (lane & 15);
    float* c1 = C + ((long)m0 + (hi << 3)) * D_HID + (nt1 << 4) + (lane & 15);
#pragma unroll
    for (int r = 0; r < 8; ++r) {
        c0[(long)r * D_HID] = acc0[r];
        c1[(long)r * D_HID] = acc1[r];
    }
}

// ---------------- conv: self-loop init acc[n][:] = dinv[n] * t[n][:] ----------------
__global__ void k_self_init(const float* __restrict__ t, const float* __restrict__ dinv,
                            float* __restrict__ acc, int n) {
    int i = blockIdx.x * blockDim.x + threadIdx.x;   // one float4 per thread
    if (i >= n * (D_HID / 4)) return;
    int node = i >> 6;                               // 64 float4 per row
    float w = dinv[node];
    float4 v = ((const float4*)t)[i];
    ((float4*)acc)[i] = make_float4(w * v.x, w * v.y, w * v.z, w * v.w);
}

// ---------------- conv: edge scatter-add, one wave per edge ----------------
__global__ void __launch_bounds__(256)
k_edge_scatter(const int* __restrict__ src, const int* __restrict__ dst,
               const float* __restrict__ t, const float* __restrict__ dinv,
               float* __restrict__ acc, int e) {
    int edge = blockIdx.x * 8 + (threadIdx.x >> 5);
    if (edge >= e) return;
    int lane = threadIdx.x & 31;
    int s = src[edge], d = dst[edge];
    float w = dinv[s];
    const float* ts = t + ((long)s << 8);
    float* ad = acc + ((long)d << 8);
#pragma unroll
    for (int i = 0; i < 8; ++i) {
        int c = lane + (i << 5);                     // coalesced 128B per step
        atomicAdd(ad + c, w * ts[c]);
    }
}

// ---------------- finalize conv1: bf16(relu(b1 + dinv[n]*acc)) ----------------
__global__ void k_finalize_relu_bf16(const float* __restrict__ acc, const float* __restrict__ dinv,
                                     const float* __restrict__ bias, __bf16* __restrict__ out, int n) {
    int i = blockIdx.x * blockDim.x + threadIdx.x;
    if (i >= n * D_HID) return;
    int node = i >> 8, d = i & 255;
    float v = bias[d] + dinv[node] * acc[i];
    out[i] = (__bf16)fmaxf(v, 0.0f);
}

// ---------------- finalize conv2: f32(b2 + dinv[n]*acc) ----------------
__global__ void k_finalize_f32(const float* __restrict__ acc, const float* __restrict__ dinv,
                               const float* __restrict__ bias, float* __restrict__ out, int n) {
    int i = blockIdx.x * blockDim.x + threadIdx.x;
    if (i >= n * D_HID) return;
    int node = i >> 8, d = i & 255;
    out[i] = bias[d] + dinv[node] * acc[i];
}

// ---------------- edge scoring: sigmoid(h2[src] . h2[dst]), one wave per edge ----------------
__global__ void __launch_bounds__(256)
k_edge_dot(const int* __restrict__ src, const int* __restrict__ dst,
           const float* __restrict__ h2, float* __restrict__ out, int e) {
    int edge = blockIdx.x * 8 + (threadIdx.x >> 5);
    if (edge >= e) return;
    int lane = threadIdx.x & 31;
    const float4* a = (const float4*)(h2 + ((long)src[edge] << 8));
    const float4* b = (const float4*)(h2 + ((long)dst[edge] << 8));
    float s = 0.0f;
#pragma unroll
    for (int i = 0; i < 2; ++i) {
        float4 va = a[lane + (i << 5)];
        float4 vb = b[lane + (i << 5)];
        s += va.x * vb.x + va.y * vb.y + va.z * vb.z + va.w * vb.w;
    }
#pragma unroll
    for (int o = 16; o > 0; o >>= 1) s += __shfl_xor(s, o, 32);
    if (lane == 0) out[edge] = 1.0f / (1.0f + __expf(-s));
}

// =====================================================================
extern "C" void kernel_launch(void* const* d_in, const int* in_sizes, int n_in,
                              void* d_out, int out_size, void* d_ws, size_t ws_size,
                              hipStream_t stream) {
    const float* x  = (const float*)d_in[0];
    const int*   ei = (const int*)d_in[1];
    const float* W1 = (const float*)d_in[2];
    const float* b1 = (const float*)d_in[3];
    const float* W2 = (const float*)d_in[4];
    const float* b2 = (const float*)d_in[5];
    float* out = (float*)d_out;

    const int N = in_sizes[0] / D_IN;       // 50000 (multiple of 16)
    const int E = in_sizes[1] / 2;          // 800000
    const int* src = ei;
    const int* dst = ei + E;

    // workspace carve-out (256B aligned)
    char* ws = (char*)d_ws;
    size_t p = 0;
    auto alloc = [&](size_t bytes) { size_t o = p; p += (bytes + 255) & ~(size_t)255; return o; };
    float*  dinv = (float*)(ws + alloc((size_t)N * 4));
    __bf16* xb   = (__bf16*)(ws + alloc((size_t)N * D_IN * 2));      // reused as h2 (f32) later
    __bf16* w1p  = (__bf16*)(ws + alloc((size_t)D_IN * D_HID * 2));
    __bf16* w2p  = (__bf16*)(ws + alloc((size_t)D_HID * D_HID * 2));
    float*  t    = (float*)(ws + alloc((size_t)N * D_HID * 4));      // GEMM output (t0, then t1)
    float*  acc  = (float*)(ws + alloc((size_t)N * D_HID * 4));      // scatter accumulator
    __bf16* h1b  = (__bf16*)(ws + alloc((size_t)N * D_HID * 2));
    float*  h2   = (float*)xb;   // N*D_IN*2 bytes == N*D_HID*4 bytes, xb dead after GEMM1

    const int T = 256;
    const int gN   = (N + T - 1) / T;
    const int gE   = (E + T - 1) / T;
    const int gEw  = (E + 7) / 8;                     // one wave per edge
    const int gXel = ((N * D_IN) + T - 1) / T;
    const int gHel = ((N * D_HID) + T - 1) / T;
    const int gH4  = ((N * (D_HID / 4)) + T - 1) / T;
    const int gMt  = N / 16;                          // M tiles for GEMM

    // ---- normalization ----
    k_deg_init<<<gN, T, 0, stream>>>(dinv, N);
    k_deg_scatter<<<gE, T, 0, stream>>>(dst, dinv, E);
    k_rsqrt_inplace<<<gN, T, 0, stream>>>(dinv, N);

    // ---- precision conversion + weight fragment packing ----
    k_cvt_bf16<<<gXel, T, 0, stream>>>(x, xb, (long)N * D_IN);
    k_pack_w<<<(D_IN * D_HID + T - 1) / T, T, 0, stream>>>(W1, w1p, D_IN, D_HID);
    k_pack_w<<<(D_HID * D_HID + T - 1) / T, T, 0, stream>>>(W2, w2p, D_HID, D_HID);

    // ---- layer 1: GEMM (WMMA bf16) + normalized scatter + ReLU ----
    k_gemm_bf16<<<gMt, T, 0, stream>>>(xb, w1p, t, D_IN);
    k_self_init<<<gH4, T, 0, stream>>>(t, dinv, acc, N);
    k_edge_scatter<<<gEw, T, 0, stream>>>(src, dst, t, dinv, acc, E);
    k_finalize_relu_bf16<<<gHel, T, 0, stream>>>(acc, dinv, b1, h1b, N);

    // ---- layer 2 ----
    k_gemm_bf16<<<gMt, T, 0, stream>>>(h1b, w2p, t, D_HID);
    k_self_init<<<gH4, T, 0, stream>>>(t, dinv, acc, N);
    k_edge_scatter<<<gEw, T, 0, stream>>>(src, dst, t, dinv, acc, E);
    k_finalize_f32<<<gHel, T, 0, stream>>>(acc, dinv, b2, h2, N);

    // ---- edge scoring ----
    k_edge_dot<<<gEw, T, 0, stream>>>(src, dst, h2, out, E);
}
